// CrossAttention_82471962018390
// MI455X (gfx1250) — matrix operand order repaired and checked by reading, not compiled, not run on hardware
//
#include <hip/hip_runtime.h>

typedef __attribute__((ext_vector_type(16))) _Float16 v16h;
typedef __attribute__((ext_vector_type(8)))  float    v8f;

#define NHEADS 8
#define DH     64
#define KS     9
#define HQ     48
#define WQ     48
#define HK     24
#define WK     24
#define NKEY   100   // 10x10 union window
#define NKPAD  112   // score N padded to 7 WMMA tiles
#define KPAD   128   // 2nd-gemm K padded to 4 chunks of 32

__device__ __forceinline__ int clampi(int x, int lo, int hi) {
    return x < lo ? lo : (x > hi ? hi : x);
}

__launch_bounds__(32)
__global__ void cross_attn_win_kernel(const float* __restrict__ q,
                                      const float* __restrict__ k,
                                      const float* __restrict__ v,
                                      float* __restrict__ out)
{
    // LDS staging (one wave per block)
    __shared__ __align__(32) _Float16 Qs[16][DH];      //  2 KB (pre-scaled by 1/8)
    __shared__ __align__(32) _Float16 Ks[NKPAD][DH];   // 14 KB
    __shared__ __align__(32) _Float16 Vt[DH][KPAD];    // 16 KB (V transposed)
    __shared__ __align__(32) _Float16 Ws[16][KPAD];    //  4 KB (softmax weights)
    __shared__ float                  Ss[16][NKPAD];   //  7 KB (scores)
    __shared__ int                    gofs[NKEY];      // clamped gi*WK+gj table

    const int tid  = threadIdx.x;         // 0..31
    const int tile = blockIdx.x;          // 0..143 : 12x12 tiles of 4x4 queries
    const int head = blockIdx.y;          // 0..7
    const int b    = blockIdx.z;          // 0..1

    const int ti0 = (tile / 12) * 4;      // first query row of tile
    const int tj0 = (tile % 12) * 4;      // first query col of tile

    // union key-window base in 24x24 space (start for the tile's first query)
    const int baseh = clampi(ti0 / 2 - 4, 0, HK - KS);
    const int basew = clampi(tj0 / 2 - 4, 0, WK - KS);
    // second-start offsets (queries in rows/cols 2..3 of the tile): 0 or 1, uniform
    const int oh1 = clampi(ti0 / 2 - 3, 0, HK - KS) - baseh;
    const int ow1 = clampi(tj0 / 2 - 3, 0, WK - KS) - basew;

    const long qcb = ((long)b * 512 + (long)head * DH) * (HQ * WQ);
    const long kcb = ((long)b * 512 + (long)head * DH) * (HK * WK);

    // ---------------- Stage 0: clamped spatial-offset table ----------------
    for (int t = tid; t < NKEY; t += 32) {
        int r = t / 10, c = t % 10;
        int gi = baseh + r; if (gi > HK - 1) gi = HK - 1;
        int gj = basew + c; if (gj > WK - 1) gj = WK - 1;
        gofs[t] = gi * WK + gj;
    }
    __syncthreads();

    // ---------------- Stage 1: global -> LDS (f32 -> f16) ------------------
    // Q tile: 16 queries x 64 dims, pre-scaled by dh^-0.5 = 0.125 (exact in f16)
    for (int idx = tid; idx < 16 * DH; idx += 32) {
        int d = idx >> 4, m = idx & 15;
        int qi = ti0 + (m >> 2), qj = tj0 + (m & 3);
        Qs[m][d] = (_Float16)(q[qcb + (long)d * (HQ * WQ) + qi * WQ + qj] * 0.125f);
    }
    // K union: 100 keys x 64 dims (row-major [key][d]); V transposed Vt[d][key]
    {
        int t = tid, d = 0;
        for (int i = 0; i < (NKEY * DH) / 32; ++i) {   // 200 iterations
            long ga = kcb + (long)d * (HK * WK) + gofs[t];
            Ks[t][d] = (_Float16)k[ga];
            Vt[d][t] = (_Float16)v[ga];
            t += 32; if (t >= NKEY) { t -= NKEY; ++d; }
        }
    }
    // zero pads: K rows 100..111, Vt cols 100..127, W cols 112..127
    for (int idx = tid; idx < (NKPAD - NKEY) * DH; idx += 32)
        Ks[NKEY + (idx >> 6)][idx & 63] = (_Float16)0.f;
    for (int idx = tid; idx < DH * (KPAD - NKEY); idx += 32)
        Vt[idx / (KPAD - NKEY)][NKEY + idx % (KPAD - NKEY)] = (_Float16)0.f;
    for (int idx = tid; idx < 16 * (KPAD - NKPAD); idx += 32)
        Ws[idx >> 4, idx / (KPAD - NKPAD)][NKPAD + idx % (KPAD - NKPAD)] = (_Float16)0.f;
    __syncthreads();

    const int lm = tid & 15;   // M row (A) or N col (B/C)
    const int hi = tid >> 4;   // half-wave select
    const int oh = hi ? oh1 : 0;   // per-lane row offset of the 9x9 mask

    // ---------------- Stage 2: scores = Q @ K^T  (16 x 112) ----------------
    // A fragment gather (contiguous k order, same as B: lanes 0-15 K=0-15, lanes 16-31 K=16-31)
    v16h qf[2];
#pragma unroll
    for (int c = 0; c < 2; ++c) {
#pragma unroll
        for (int e = 0; e < 16; ++e) {
            int j  = e >> 1;
            int kl = hi * 16 + e; (void)j;
            qf[c][e] = Qs[lm][c * 32 + kl];
        }
    }

#pragma unroll
    for (int nt = 0; nt < 7; ++nt) {
        v8f acc = {};
#pragma unroll
        for (int c = 0; c < 2; ++c) {
            // B fragment: lane holds key t = nt*16+lm, 16 contiguous halves of d
            const v16h bf = *(const v16h*)&Ks[nt * 16 + lm][c * 32 + hi * 16];
            acc = __builtin_amdgcn_wmma_f32_16x16x32_f16(
                      false, qf[c], false, bf, (short)0, acc, false, false);
        }
        // C layout: lane holds N = lm, VGPR g -> M = g + 8*hi
        const int t  = nt * 16 + lm;
        const int r  = t / 10;
        const int cc = t - r * 10;
        const int rok = (t < NKEY) & (r >= oh) & (r <= oh + 8);
        const int vA  = rok & (cc <= 8);                       // cols 0,1: ow=0
        const int vB  = rok & (cc >= ow1) & (cc <= ow1 + 8);   // cols 2,3: ow=ow1
#pragma unroll
        for (int g = 0; g < 8; ++g) {
            int valid = (g & 2) ? vB : vA;        // col = g&3 -> compile-time pick
            Ss[g + 8 * hi][t] = valid ? acc[g] : -1e30f;
        }
    }
    __syncthreads();

    // ---------------- Stage 3: softmax over 81 valid keys per row ----------
    if (tid < 16) {
        float mx = -1e30f;
        for (int t = 0; t < NKPAD; ++t) mx = fmaxf(mx, Ss[tid][t]);
        float sum = 0.f;
        for (int t = 0; t < NKPAD; ++t) {
            float e = __expf(Ss[tid][t] - mx);    // masked -> underflow to 0
            Ss[tid][t] = e;
            sum += e;
        }
        float inv = 1.f / sum;
        for (int t = 0; t < NKPAD; ++t)
            Ws[tid][t] = (_Float16)(Ss[tid][t] * inv);
    }
    __syncthreads();

    // ---------------- Stage 4: out = W @ V  (16 x 64) ----------------------
    v16h wf[4];
#pragma unroll
    for (int c = 0; c < 4; ++c) {
#pragma unroll
        for (int e = 0; e < 16; ++e) {
            int j  = e >> 1;
            int kl = hi * 16 + e; (void)j;
            wf[c][e] = Ws[lm][c * 32 + kl];
        }
    }

#pragma unroll
    for (int nt = 0; nt < 4; ++nt) {
        v8f acc = {};
#pragma unroll
        for (int c = 0; c < 4; ++c) {
            // B fragment: lane holds out-dim d = nt*16+lm, contiguous key halves
            const v16h bf = *(const v16h*)&Vt[nt * 16 + lm][c * 32 + hi * 16];
            acc = __builtin_amdgcn_wmma_f32_16x16x32_f16(
                      false, wf[c], false, bf, (short)0, acc, false, false);
        }
        // acc[0..3] = 4 consecutive pixels of row ti0+2*hi, acc[4..7] = next row
        const int d = nt * 16 + lm;
        float* obase = out + qcb + (long)d * (HQ * WQ) + (ti0 + 2 * hi) * WQ + tj0;
        float4 o0 = {acc[0], acc[1], acc[2], acc[3]};
        float4 o1 = {acc[4], acc[5], acc[6], acc[7]};
        *(float4*)(obase)      = o0;
        *(float4*)(obase + WQ) = o1;
    }
}

extern "C" void kernel_launch(void* const* d_in, const int* in_sizes, int n_in,
                              void* d_out, int out_size, void* d_ws, size_t ws_size,
                              hipStream_t stream) {
    const float* q = (const float*)d_in[0];   // (2, 512, 48, 48)
    const float* k = (const float*)d_in[1];   // (2, 512, 24, 24)
    const float* v = (const float*)d_in[2];   // (2, 512, 24, 24)
    float* out = (float*)d_out;               // (2, 512, 48, 48)

    dim3 grid(144, NHEADS, 2);   // 12x12 query tiles, 8 heads, batch 2
    dim3 block(32);              // one wave32 per (tile, head, batch)
    cross_attn_win_kernel<<<grid, block, 0, stream>>>(q, k, v, out);
}
